// LSTM_25142738550794
// MI455X (gfx1250) — compile-verified, matrix operand order checked by reading, not run on hardware
//
#include <hip/hip_runtime.h>
#include <hip/hip_bf16.h>

// ---------------- problem constants (from reference setup) ----------------
#define BB   32
#define SS   128
#define II   1024
#define HH   1024
#define VV   32000
#define CTX  40              // context_seq (python scalar; fixed by setup)
#define SOUT (SS - CTX)      // 88
#define G4   (4 * HH)        // 4096 gate width

typedef __attribute__((ext_vector_type(16))) __bf16 v16bf;
typedef __attribute__((ext_vector_type(8)))  __bf16 v8bf;
typedef __attribute__((ext_vector_type(8)))  float  v8f;
typedef __attribute__((ext_vector_type(4)))  int    v4i;

union BF16x16 { v16bf v; v8bf h[2]; };

// ---- gfx1250 async global->LDS path (guarded; falls back to reg bounce) ----
#ifdef __has_builtin
#if __has_builtin(__builtin_amdgcn_global_load_async_to_lds_b128)
#define ASYNC_LDS 1
#endif
#endif
#ifndef ASYNC_LDS
#define ASYNC_LDS 0
#endif

#if ASYNC_LDS
typedef __attribute__((address_space(1))) v4i* gv4i_p;
typedef __attribute__((address_space(3))) v4i* lv4i_p;

__device__ __forceinline__ void async_cp16(const __bf16* g, __bf16* l) {
    __builtin_amdgcn_global_load_async_to_lds_b128(
        (gv4i_p)g, (lv4i_p)l, 0, 0);
}
__device__ __forceinline__ void wait_async0() {
#if __has_builtin(__builtin_amdgcn_s_wait_asynccnt)
    __builtin_amdgcn_s_wait_asynccnt(0);
#else
    asm volatile("s_wait_asynccnt 0" ::: "memory");
#endif
}
#endif

__device__ __forceinline__ v8f wmma_bf16(const BF16x16& a, const BF16x16& b, v8f c) {
    return __builtin_amdgcn_wmma_f32_16x16x32_bf16(false, a.v, false, b.v,
                                                   (short)0, c, false, false);
}

// ===========================================================================
// Block-tiled GEMM through LDS:  C[M x N] = A[M x K] * W[N x K]^T + bias
// Block = 256 thr = 8 waves (2 x 4).  Block tile 64 x 256, wave tile 32 x 64,
// LDS K-stage 64 (two 32-deep WMMA sub-steps), double-buffered.
// LDS rows padded to 72 elems (144 B = 36 words): lane L of a fragment read
// starts at bank 36*L mod 64 -> sixteen disjoint 4-word runs covering all 64
// banks, i.e. conflict-free ds_load_b128 (vs 8-way conflicts at 128 B stride).
// ===========================================================================
#define TBM 64
#define TBN 256
#define TKK 64
#define TKP (TKK + 8)        // padded LDS row stride (elements)

__global__ __launch_bounds__(256) void k_gemm_tiled(
    const __bf16* __restrict__ A, const __bf16* __restrict__ W,
    float* __restrict__ C,
    const float* __restrict__ bias0, const float* __restrict__ bias1,
    int M, int N, int K)
{
    __shared__ __bf16 sA[2][TBM * TKP];     // 2 * 9 KB
    __shared__ __bf16 sB[2][TBN * TKP];     // 2 * 36 KB

    const int tid  = threadIdx.x;
    const int lane = tid & 31;
    const int wave = tid >> 5;
    const int wm   = wave >> 2;      // 0..1
    const int wn   = wave & 3;       // 0..3
    const int half = lane >> 4;
    const int l16  = lane & 15;
    const int kb   = half * 8;

    const int nBlocksN = N / TBN;
    const int row0 = (blockIdx.x / nBlocksN) * TBM;
    const int col0 = (blockIdx.x % nBlocksN) * TBN;

    // accumulators, init with bias
    v8f acc[2][4];
#pragma unroll
    for (int t = 0; t < 4; ++t) {
        const int col = col0 + wn * 64 + t * 16 + l16;
        float bv = 0.0f;
        if (bias0) bv += bias0[col];
        if (bias1) bv += bias1[col];
#pragma unroll
        for (int r = 0; r < 8; ++r) { acc[0][t][r] = bv; acc[1][t][r] = bv; }
    }

    const int kTiles = K / TKK;

#if ASYNC_LDS
    auto load_tile = [&](int kt, int buf) {
        // A: 64x64 = 512 16B-chunks -> 2 per thread
#pragma unroll
        for (int i = 0; i < 2; ++i) {
            const int ch = tid + i * 256;
            const int ar = ch >> 3, ac = (ch & 7) * 8;
            async_cp16(A + (size_t)(row0 + ar) * K + kt * TKK + ac,
                       &sA[buf][ar * TKP + ac]);
        }
        // B: 256x64 = 2048 16B-chunks -> 8 per thread
#pragma unroll
        for (int i = 0; i < 8; ++i) {
            const int ch = tid + i * 256;
            const int br = ch >> 3, bc = (ch & 7) * 8;
            async_cp16(W + (size_t)(col0 + br) * K + kt * TKK + bc,
                       &sB[buf][br * TKP + bc]);
        }
    };
#else
    v8bf ra[2]; v8bf rb[8];
    auto load_regs = [&](int kt) {
#pragma unroll
        for (int i = 0; i < 2; ++i) {
            const int ch = tid + i * 256;
            const int ar = ch >> 3, ac = (ch & 7) * 8;
            ra[i] = *(const v8bf*)(A + (size_t)(row0 + ar) * K + kt * TKK + ac);
        }
#pragma unroll
        for (int i = 0; i < 8; ++i) {
            const int ch = tid + i * 256;
            const int br = ch >> 3, bc = (ch & 7) * 8;
            rb[i] = *(const v8bf*)(W + (size_t)(col0 + br) * K + kt * TKK + bc);
        }
    };
    auto store_regs = [&](int buf) {
#pragma unroll
        for (int i = 0; i < 2; ++i) {
            const int ch = tid + i * 256;
            const int ar = ch >> 3, ac = (ch & 7) * 8;
            *(v8bf*)&sA[buf][ar * TKP + ac] = ra[i];
        }
#pragma unroll
        for (int i = 0; i < 8; ++i) {
            const int ch = tid + i * 256;
            const int br = ch >> 3, bc = (ch & 7) * 8;
            *(v8bf*)&sB[buf][br * TKP + bc] = rb[i];
        }
    };
#endif

    auto compute = [&](int buf) {
#pragma unroll
        for (int ks = 0; ks < TKK; ks += 32) {
            BF16x16 afr[2], bfr[4];
#pragma unroll
            for (int a = 0; a < 2; ++a) {
                const __bf16* p = &sA[buf][(wm * 32 + a * 16 + l16) * TKP + ks + kb];
                afr[a].h[0] = *(const v8bf*)(p);
                afr[a].h[1] = *(const v8bf*)(p + 16);
            }
#pragma unroll
            for (int t = 0; t < 4; ++t) {
                const __bf16* p = &sB[buf][(wn * 64 + t * 16 + l16) * TKP + ks + kb];
                bfr[t].h[0] = *(const v8bf*)(p);
                bfr[t].h[1] = *(const v8bf*)(p + 16);
            }
#pragma unroll
            for (int t = 0; t < 4; ++t) {
                acc[0][t] = wmma_bf16(afr[0], bfr[t], acc[0][t]);
                acc[1][t] = wmma_bf16(afr[1], bfr[t], acc[1][t]);
            }
        }
    };

#if ASYNC_LDS
    load_tile(0, 0);
    wait_async0();
    __syncthreads();
    for (int kt = 0; kt < kTiles; ++kt) {
        const int buf = kt & 1;
        const bool more = (kt + 1) < kTiles;
        if (more) load_tile(kt + 1, buf ^ 1);   // async into other buffer
        compute(buf);
        if (more) wait_async0();                // own-wave async completion
        __syncthreads();                        // cross-wave LDS visibility
    }
#else
    load_regs(0);
    store_regs(0);
    __syncthreads();
    for (int kt = 0; kt < kTiles; ++kt) {
        const int buf = kt & 1;
        const bool more = (kt + 1) < kTiles;
        if (more) load_regs(kt + 1);
        compute(buf);
        __syncthreads();
        if (more) {
            store_regs(buf ^ 1);
            __syncthreads();
        }
    }
#endif

    // store C
#pragma unroll
    for (int a = 0; a < 2; ++a) {
        const int r0 = row0 + wm * 32 + a * 16 + half * 8;
#pragma unroll
        for (int t = 0; t < 4; ++t) {
            const int col = col0 + wn * 64 + t * 16 + l16;
            float* cp = C + (size_t)r0 * N + col;
#pragma unroll
            for (int r = 0; r < 8; ++r) cp[(size_t)r * N] = acc[a][t][r];
        }
    }
}

// ===========================================================================
// Fused LSTM step: gates = x_gates[s] + h_prev @ W_hh^T, then cell update,
// entirely in registers. Wave owns (16 batch rows) x (16 hidden cols) and
// computes all four gate tiles (rows j, H+j, 2H+j, 3H+j of W_hh).
// Grid: 2 m-tiles x 64 j-tiles = 128 waves = 16 blocks.
// ===========================================================================
__device__ __forceinline__ float sigmoidf_(float x) {
    return 1.0f / (1.0f + __expf(-x));
}

__global__ __launch_bounds__(256) void k_step(
    const __bf16* __restrict__ h_in, const __bf16* __restrict__ whh,
    const float* __restrict__ xg,          // (B, 4H) for this step, has biases
    float* __restrict__ c, __bf16* __restrict__ h_out,
    __bf16* __restrict__ hs, int s)
{
    const long wid  = ((long)blockIdx.x * blockDim.x + threadIdx.x) >> 5;
    const int  tm   = (int)(wid >> 6);      // 0..1
    const int  j0   = ((int)(wid & 63)) * 16;
    const int  lane = threadIdx.x & 31;
    const int  half = lane >> 4;
    const int  l16  = lane & 15;
    const int  kb   = half * 8;
    const int  row0 = tm * 16;

    // init accumulators from precomputed input gates (biases included)
    v8f acc[4];
#pragma unroll
    for (int g = 0; g < 4; ++g) {
        const float* gp = xg + (size_t)(row0 + half * 8) * G4 + g * HH + j0 + l16;
#pragma unroll
        for (int r = 0; r < 8; ++r) acc[g][r] = gp[(size_t)r * G4];
    }

    if (s > 0) {
        BF16x16 a_cur, a_nxt, b_cur[4], b_nxt[4];
        auto loadA = [&](int kk, BF16x16& f) {
            const __bf16* p = h_in + (size_t)(row0 + l16) * HH + kk + kb;
            f.h[0] = *(const v8bf*)(p); f.h[1] = *(const v8bf*)(p + 16);
        };
        auto loadB = [&](int kk, int g, BF16x16& f) {
            const __bf16* p = whh + (size_t)(g * HH + j0 + l16) * HH + kk + kb;
            f.h[0] = *(const v8bf*)(p); f.h[1] = *(const v8bf*)(p + 16);
        };
        loadA(0, a_cur);
#pragma unroll
        for (int g = 0; g < 4; ++g) loadB(0, g, b_cur[g]);

        for (int kk = 0; kk < HH; kk += 32) {
            const bool more = (kk + 32) < HH;
            if (more) {                              // prefetch next fragments
                loadA(kk + 32, a_nxt);
#pragma unroll
                for (int g = 0; g < 4; ++g) loadB(kk + 32, g, b_nxt[g]);
            }
#pragma unroll
            for (int g = 0; g < 4; ++g) acc[g] = wmma_bf16(a_cur, b_cur[g], acc[g]);
            if (more) {
                a_cur = a_nxt;
#pragma unroll
                for (int g = 0; g < 4; ++g) b_cur[g] = b_nxt[g];
            }
        }
    }

    // cell update in registers: acc[0]=i, acc[1]=f, acc[2]=g, acc[3]=o
#pragma unroll
    for (int r = 0; r < 8; ++r) {
        const int b = row0 + half * 8 + r;
        const int j = j0 + l16;
        const float ig = sigmoidf_(acc[0][r]);
        const float fg = sigmoidf_(acc[1][r]);
        const float gg = tanhf(acc[2][r]);
        const float og = sigmoidf_(acc[3][r]);
        const float cp = (s == 0) ? 0.0f : c[(size_t)b * HH + j];
        const float cn = fg * cp + ig * gg;
        const float hv = og * tanhf(cn);
        c[(size_t)b * HH + j] = cn;
        h_out[(size_t)b * HH + j] = (__bf16)hv;
        if (s >= CTX)
            hs[((size_t)b * SOUT + (s - CTX)) * HH + j] = (__bf16)hv;
    }
}

// ----------------------------- small kernels --------------------------------

__global__ __launch_bounds__(256) void k_f32_to_bf16(
    const float* __restrict__ src, __bf16* __restrict__ dst, long n)
{
    long i = (long)blockIdx.x * blockDim.x + threadIdx.x;
    const long stride = (long)gridDim.x * blockDim.x;
    for (; i < n; i += stride) dst[i] = (__bf16)src[i];
}

__global__ __launch_bounds__(256) void k_embed(
    const int* __restrict__ ids, const float* __restrict__ emb,
    __bf16* __restrict__ x)
{
    const int m = blockIdx.x;            // m = s*B + b
    const int s = m / BB;
    const int b = m % BB;
    const int id = ids[b * SS + s];
    const float* e = emb + (size_t)id * II;
    __bf16* xr = x + (size_t)m * II;
    for (int i = threadIdx.x; i < II; i += blockDim.x) xr[i] = (__bf16)e[i];
}

// ----------------------------- host side -----------------------------------

static inline size_t align256(size_t x) { return (x + 255) & ~(size_t)255; }

extern "C" void kernel_launch(void* const* d_in, const int* in_sizes, int n_in,
                              void* d_out, int out_size, void* d_ws, size_t ws_size,
                              hipStream_t stream)
{
    const int*   ids   = (const int*)  d_in[0];
    const float* emb   = (const float*)d_in[1];
    const float* w_ih  = (const float*)d_in[2];
    const float* w_hh  = (const float*)d_in[3];
    const float* b_ih  = (const float*)d_in[4];
    const float* b_hh  = (const float*)d_in[5];
    const float* w_out = (const float*)d_in[6];
    const float* b_out = (const float*)d_in[7];
    // d_in[8] = context_seq (device scalar) — fixed at 40 per setup.

    float* out = (float*)d_out;        // (B*SOUT, V) f32

    // ---- workspace partition (~164 MB) ----
    char* p = (char*)d_ws;
    __bf16* x_bf   = (__bf16*)p; p += align256((size_t)SS * BB * II * 2);
    __bf16* wih_bf = (__bf16*)p; p += align256((size_t)G4 * II * 2);
    __bf16* whh_bf = (__bf16*)p; p += align256((size_t)G4 * HH * 2);
    __bf16* wout_bf= (__bf16*)p; p += align256((size_t)VV * HH * 2);
    float*  x_g    = (float*) p; p += align256((size_t)SS * BB * G4 * 4);
    float*  c_st   = (float*) p; p += align256((size_t)BB * HH * 4);
    __bf16* h_st0  = (__bf16*)p; p += align256((size_t)BB * HH * 2);
    __bf16* h_st1  = (__bf16*)p; p += align256((size_t)BB * HH * 2);
    __bf16* hs_bf  = (__bf16*)p; p += align256((size_t)BB * SOUT * HH * 2);
    (void)ws_size; (void)in_sizes; (void)n_in; (void)out_size;

    // ---- 1) weight conversions to bf16 ----
    k_f32_to_bf16<<<2048, 256, 0, stream>>>(w_ih,  wih_bf,  (long)G4 * II);
    k_f32_to_bf16<<<2048, 256, 0, stream>>>(w_hh,  whh_bf,  (long)G4 * HH);
    k_f32_to_bf16<<<4096, 256, 0, stream>>>(w_out, wout_bf, (long)VV * HH);

    // ---- 2) embedding gather -> (S*B, I) bf16 ----
    k_embed<<<SS * BB, 256, 0, stream>>>(ids, emb, x_bf);

    // ---- 3) x_gates = X * W_ih^T + b_ih + b_hh   (4096 x 4096) ----
    {
        const int M = SS * BB, N = G4;
        const int blocks = (M / TBM) * (N / TBN);          // 64 * 16
        k_gemm_tiled<<<blocks, 256, 0, stream>>>(
            x_bf, wih_bf, x_g, b_ih, b_hh, M, N, II);
    }

    // ---- 4) sequential LSTM scan: one fused kernel per step ----
    for (int s = 0; s < SS; ++s) {
        float* gates_s = x_g + (size_t)s * BB * G4;
        const __bf16* hin  = (s & 1) ? h_st1 : h_st0;
        __bf16*       hout = (s & 1) ? h_st0 : h_st1;
        k_step<<<16, 256, 0, stream>>>(hin, whh_bf, gates_s,
                                       c_st, hout, hs_bf, s);
    }

    // ---- 5) logits = hs * W_out^T + b_out   (2816 x 32000) ----
    {
        const int M = BB * SOUT, N = VV;
        const int blocks = (M / TBM) * (N / TBN);          // 44 * 125
        k_gemm_tiled<<<blocks, 256, 0, stream>>>(
            hs_bf, wout_bf, out, b_out, nullptr, M, N, HH);
    }
}
